// DlrmSmall_74758200754619
// MI455X (gfx1250) — matrix operand order repaired and checked by reading, not compiled, not run on hardware
//
#include <hip/hip_runtime.h>
#include <hip/hip_bf16.h>
#include <stdint.h>

#define BATCH  16384
#define VOCABM (4194304 - 1)   // VOCAB is a power of two -> idx % VOCAB == idx & VOCABM

typedef __attribute__((ext_vector_type(16))) __bf16 v16bf;
typedef __attribute__((ext_vector_type(8)))  float  v8f;

union Frag {
  uint4    q[2];
  uint32_t u[8];
  v16bf    v;
};

__device__ __forceinline__ uint16_t f2bf(float x) {
  uint32_t u = __float_as_uint(x);
  u += 0x7FFFu + ((u >> 16) & 1u);           // round-to-nearest-even
  return (uint16_t)(u >> 16);
}
__device__ __forceinline__ uint32_t pack2(float a, float b) {
  return (uint32_t)f2bf(a) | ((uint32_t)f2bf(b) << 16);
}
__device__ __forceinline__ float bf2f(uint32_t h) {
  return __uint_as_float(h << 16);
}

// ---------------------------------------------------------------------------
// f32 -> padded bf16 input (B x 13 -> B x 32, zero padded)
// ---------------------------------------------------------------------------
__global__ __launch_bounds__(256) void pad_xdense(const float* __restrict__ x,
                                                  uint16_t* __restrict__ xp) {
  int idx = blockIdx.x * 256 + threadIdx.x;
  int r = idx >> 5, c = idx & 31;
  float v = (c < 13) ? x[r * 13 + c] : 0.0f;
  xp[idx] = f2bf(v);
}

// ---------------------------------------------------------------------------
// f32 W (K x N) -> bf16 W^T (N x Kpad), zero padded along K
// ---------------------------------------------------------------------------
__global__ __launch_bounds__(256) void convert_wT(const float* __restrict__ W,
                                                  uint16_t* __restrict__ WT,
                                                  int K, int N, int Kpad) {
  int idx = blockIdx.x * 256 + threadIdx.x;
  if (idx >= N * Kpad) return;
  int n = idx / Kpad, k = idx - n * Kpad;
  float v = (k < K) ? W[(size_t)k * N + n] : 0.0f;
  WT[idx] = f2bf(v);
}

// ---------------------------------------------------------------------------
// Register-blocked GEMM: each wave computes a 32x64 output tile
// (2 M-subtiles x 4 N-subtiles) with v_wmma_f32_16x16x32_bf16.
// Per K-step: 2 A-fragments + 4 B-fragments (6 KB L2 traffic) feed 8 WMMAs
// (131K FLOP) -> ~22 FLOP/byte from L2, vs 8 FLOP/byte unblocked.
// A-fragment per ISA 16-bit A(16x32) layout, B-fragment per B(32x16) layout.
// ---------------------------------------------------------------------------
__global__ __launch_bounds__(256) void gemm_bf16_wmma(
    const uint16_t* __restrict__ A, const uint16_t* __restrict__ WT,
    const float* __restrict__ bias, uint16_t* __restrict__ Out,
    int M, int N, int K, int relu)
{
  const int lane = threadIdx.x & 31;
  const int wave = threadIdx.x >> 5;
  const int tile = blockIdx.x * 8 + wave;
  const int ntn  = N >> 6;               // 64-wide N blocks
  const int tm   = tile / ntn;           // 32-tall M blocks
  const int tn   = tile - tm * ntn;
  if (tm >= (M >> 5)) return;            // wave-uniform guard
  const int rowl = lane & 15;
  const int half = lane >> 4;

  const uint16_t* arow0 = A  + (size_t)((tm << 5) + rowl) * K;
  const uint16_t* brow0 = WT + (size_t)((tn << 6) + rowl) * K;

  v8f acc[2][4] = {};
  for (int kc = 0; kc < K; kc += 32) {
    Frag fa[2];
    #pragma unroll
    for (int m = 0; m < 2; ++m) {
      // A: VGPR v holds K-pair (v&3)*2 + 8*half (+16 for v>=4) -> two b128 loads
      const uint16_t* ar = arow0 + (size_t)(m * 16) * K + kc;
      fa[m].q[0] = *(const uint4*)(ar + half * 8);
      fa[m].q[1] = *(const uint4*)(ar + half * 8 + 16);
    }
    #pragma unroll
    for (int n = 0; n < 4; ++n) {
      // B: VGPR v holds K-pair (v&3)*2 + 8*half (+16 for v>=4), column = lane&15 -> two b128 loads
      const uint16_t* br = brow0 + (size_t)(n * 16) * K + kc;
      Frag fb;
      fb.q[0] = *(const uint4*)(br + half * 8);
      fb.q[1] = *(const uint4*)(br + half * 8 + 16);
      #pragma unroll
      for (int m = 0; m < 2; ++m)
        acc[m][n] = __builtin_amdgcn_wmma_f32_16x16x32_bf16(
            false, fa[m].v, false, fb.v, (short)0, acc[m][n], false, false);
    }
  }

  #pragma unroll
  for (int n = 0; n < 4; ++n) {
    const float bv = bias[(tn << 6) + n * 16 + rowl];   // N = lane&15 (both halves)
    #pragma unroll
    for (int m = 0; m < 2; ++m) {
      uint16_t* orow = Out + (size_t)((tm << 5) + m * 16 + half * 8) * N
                           + (tn << 6) + n * 16 + rowl;
      #pragma unroll
      for (int r = 0; r < 8; ++r) {                     // VGPR r -> M = r (+8 hi half)
        float x = acc[m][n][r] + bv;
        if (relu) x = fmaxf(x, 0.0f);
        orow[(size_t)r * N] = f2bf(x);
      }
    }
  }
}

// ---------------------------------------------------------------------------
// Fused embedding gather + pairwise interaction + top-input assembly.
// One wave per sample: stage 32x128 bf16 stack in LDS, 3 WMMA tiles of the
// Gram matrix, scatter 378 upper-triangular values into top_in[128..505].
// top_in layout per row (512): [bot(128) | triu(378) | zeros(6)]
// ---------------------------------------------------------------------------
__global__ __launch_bounds__(256) void interact(
    const uint16_t* __restrict__ bot, const int* __restrict__ xcat,
    const float* __restrict__ table, uint16_t* __restrict__ topin)
{
  __shared__ __align__(16) uint32_t smem[8 * 2048];  // 8 waves * (32 rows * 64 dwords)
  const int lane = threadIdx.x & 31;
  const int wave = threadIdx.x >> 5;
  const size_t s = (size_t)blockIdx.x * 8 + wave;
  uint32_t* S = smem + wave * 2048;

  // row 0: bottom-MLP output (already bf16) -> LDS and straight into top_in[0..127]
  const uint32_t* b32 = (const uint32_t*)(bot + s * 128);
  uint32_t* t32 = (uint32_t*)(topin + s * 512);
  uint32_t d0 = b32[lane * 2], d1 = b32[lane * 2 + 1];
  S[lane * 2] = d0;  S[lane * 2 + 1] = d1;
  t32[lane * 2] = d0; t32[lane * 2 + 1] = d1;

  // rows 1..26: gather embedding rows (f32) and convert to bf16
  #pragma unroll 1
  for (int r = 1; r <= 26; ++r) {
    int idx = xcat[s * 26 + (r - 1)] & VOCABM;
    const float4 f = ((const float4*)(table + (size_t)idx * 128))[lane];
    S[r * 64 + lane * 2]     = pack2(f.x, f.y);
    S[r * 64 + lane * 2 + 1] = pack2(f.z, f.w);
  }
  // rows 27..31: zero pad (so the 32x32 Gram is exact on the 27x27 block)
  for (int i = lane; i < 320; i += 32) S[27 * 64 + i] = 0;
  // zero top_in[506..511]
  if (lane < 3) t32[253 + lane] = 0;
  __syncthreads();

  const int rowl = lane & 15, half = lane >> 4;
  #pragma unroll
  for (int t = 0; t < 3; ++t) {           // tiles (0,0), (0,1), (1,1)
    const int mi = (t == 2) ? 1 : 0;
    const int ni = (t == 0) ? 0 : 1;
    v8f acc = {};
    #pragma unroll
    for (int kc = 0; kc < 128; kc += 32) {
      Frag fa, fb;
      const uint32_t* ar = S + (mi * 16 + rowl) * 64;
      const uint32_t* br = S + (ni * 16 + rowl) * 64;   // B[k][n] = S[n][k]
      fa.q[0] = *(const uint4*)(ar + ((kc + half * 8) >> 1));
      fa.q[1] = *(const uint4*)(ar + ((kc + half * 8 + 16) >> 1));
      fb.q[0] = *(const uint4*)(br + ((kc + half * 8) >> 1));
      fb.q[1] = *(const uint4*)(br + ((kc + half * 8 + 16) >> 1));
      acc = __builtin_amdgcn_wmma_f32_16x16x32_bf16(false, fa.v, false, fb.v,
                                                    (short)0, acc, false, false);
    }
    #pragma unroll
    for (int r = 0; r < 8; ++r) {
      const int i = mi * 16 + r + half * 8;
      const int j = ni * 16 + rowl;
      if (i <= j && j <= 26) {
        const int off = i * 27 - (i * (i - 1)) / 2 + (j - i);  // triu index
        topin[s * 512 + 128 + off] = f2bf(acc[r]);
      }
    }
  }
}

// ---------------------------------------------------------------------------
// Final 256 -> 1 layer (N=1: plain per-thread dot product, f32 out)
// ---------------------------------------------------------------------------
__global__ __launch_bounds__(256) void final_layer(
    const uint16_t* __restrict__ T3, const float* __restrict__ w,
    const float* __restrict__ b, float* __restrict__ out)
{
  int s = blockIdx.x * 256 + threadIdx.x;
  if (s >= BATCH) return;
  const uint16_t* row = T3 + (size_t)s * 256;
  float acc = b[0];
  for (int k = 0; k < 256; k += 2) {
    uint32_t p = *(const uint32_t*)(row + k);
    acc = fmaf(bf2f(p & 0xFFFFu), w[k], acc);
    acc = fmaf(bf2f(p >> 16), w[k + 1], acc);
  }
  out[s] = acc;
}

// ---------------------------------------------------------------------------
extern "C" void kernel_launch(void* const* d_in, const int* in_sizes, int n_in,
                              void* d_out, int out_size, void* d_ws, size_t ws_size,
                              hipStream_t stream)
{
  (void)in_sizes; (void)n_in; (void)out_size; (void)ws_size;
  const float* x_dense = (const float*)d_in[0];
  const int*   x_cat   = (const int*)  d_in[1];
  const float* table   = (const float*)d_in[2];
  const float* w_bot0  = (const float*)d_in[3];  const float* b_bot0 = (const float*)d_in[4];
  const float* w_bot1  = (const float*)d_in[5];  const float* b_bot1 = (const float*)d_in[6];
  const float* w_bot2  = (const float*)d_in[7];  const float* b_bot2 = (const float*)d_in[8];
  const float* w_top0  = (const float*)d_in[9];  const float* b_top0 = (const float*)d_in[10];
  const float* w_top1  = (const float*)d_in[11]; const float* b_top1 = (const float*)d_in[12];
  const float* w_top2  = (const float*)d_in[13]; const float* b_top2 = (const float*)d_in[14];
  const float* w_top3  = (const float*)d_in[15]; const float* b_top3 = (const float*)d_in[16];
  const float* w_top4  = (const float*)d_in[17]; const float* b_top4 = (const float*)d_in[18];

  uintptr_t base = (uintptr_t)d_ws;
  size_t off = 0;
  auto alloc = [&](size_t elems) -> uint16_t* {
    uint16_t* p = (uint16_t*)(base + off);
    off += (elems * 2 + 255) & ~(size_t)255;
    return p;
  };
  uint16_t* xp    = alloc((size_t)BATCH * 32);
  uint16_t* wb0   = alloc((size_t)512 * 32);
  uint16_t* wb1   = alloc((size_t)256 * 512);
  uint16_t* wb2   = alloc((size_t)128 * 256);
  uint16_t* wt0   = alloc((size_t)1024 * 512);
  uint16_t* wt1   = alloc((size_t)1024 * 1024);
  uint16_t* wt2   = alloc((size_t)512 * 1024);
  uint16_t* wt3   = alloc((size_t)256 * 512);
  uint16_t* h512  = alloc((size_t)BATCH * 512);   // h0, later reused as t2
  uint16_t* h256  = alloc((size_t)BATCH * 256);   // h1, later reused as t3
  uint16_t* botb  = alloc((size_t)BATCH * 128);
  uint16_t* topin = alloc((size_t)BATCH * 512);
  uint16_t* t0    = alloc((size_t)BATCH * 1024);
  uint16_t* t1    = alloc((size_t)BATCH * 1024);

  pad_xdense<<<(BATCH * 32) / 256, 256, 0, stream>>>(x_dense, xp);
  auto cvt = [&](const float* W, uint16_t* WT, int K, int N, int Kpad) {
    int total = N * Kpad;
    convert_wT<<<(total + 255) / 256, 256, 0, stream>>>(W, WT, K, N, Kpad);
  };
  cvt(w_bot0, wb0, 13,  512, 32);
  cvt(w_bot1, wb1, 512, 256, 512);
  cvt(w_bot2, wb2, 256, 128, 256);
  cvt(w_top0, wt0, 506, 1024, 512);
  cvt(w_top1, wt1, 1024, 1024, 1024);
  cvt(w_top2, wt2, 1024, 512, 1024);
  cvt(w_top3, wt3, 512, 256, 512);

  auto gemm = [&](const uint16_t* A, const uint16_t* WT, const float* bias,
                  uint16_t* Out, int M, int N, int K, int relu) {
    int tiles = (M >> 5) * (N >> 6);          // 32x64 tiles; always a multiple of 8 here
    gemm_bf16_wmma<<<tiles / 8, 256, 0, stream>>>(A, WT, bias, Out, M, N, K, relu);
  };
  // bottom MLP
  gemm(xp,   wb0, b_bot0, h512, BATCH, 512, 32,  1);
  gemm(h512, wb1, b_bot1, h256, BATCH, 256, 512, 1);
  gemm(h256, wb2, b_bot2, botb, BATCH, 128, 256, 1);
  // fused gather + interaction + top-input assembly
  interact<<<BATCH / 8, 256, 0, stream>>>(botb, x_cat, table, topin);
  // top MLP
  gemm(topin, wt0, b_top0, t0,   BATCH, 1024, 512,  1);
  gemm(t0,    wt1, b_top1, t1,   BATCH, 1024, 1024, 1);
  gemm(t1,    wt2, b_top2, h512, BATCH, 512,  1024, 1);
  gemm(h512,  wt3, b_top3, h256, BATCH, 256,  512,  1);
  final_layer<<<BATCH / 256, 256, 0, stream>>>(h256, w_top4, b_top4, (float*)d_out);
}
